// EquivariantProteinGNN_45552423141948
// MI455X (gfx1250) — compile-verified, matrix operand order checked by reading, not run on hardware
//
#include <hip/hip_runtime.h>
#include <hip/hip_bf16.h>
#include <math.h>

// ---------------- problem constants (match reference) ----------------
#define NN   20000
#define EE   200000
#define GG   32
#define DD   384
#define HH   12
#define CC   32
#define LL   6
#define NRBF 100

#define SBS  (DD + 8)      // padded LDS row stride (halves): 784B -> conflict-free

typedef _Float16 v16h __attribute__((ext_vector_type(16)));
typedef _Float16 v8h  __attribute__((ext_vector_type(8)));
typedef float    v8f  __attribute__((ext_vector_type(8)));

// ---------------- helpers ----------------
__device__ __forceinline__ float silu_f(float x) { return x / (1.0f + __expf(-x)); }

// order-preserving float <-> uint mapping for atomicMax-based segment max
__device__ __forceinline__ unsigned f2o(float f) {
    unsigned u = __float_as_uint(f);
    return (u & 0x80000000u) ? ~u : (u | 0x80000000u);
}
__device__ __forceinline__ float o2f(unsigned u) {
    unsigned v = (u & 0x80000000u) ? (u & 0x7FFFFFFFu) : ~u;
    return __uint_as_float(v);
}

// ---------------- weight prep: W[K][N] f32 -> Bt[N][K] f16 ----------------
__global__ __launch_bounds__(256) void k_prep_bt(
    const float* __restrict__ W, _Float16* __restrict__ Bt)
{
    const size_t i = (size_t)blockIdx.x * 256 + threadIdx.x;   // over N*K, N==K==DD
    if (i >= (size_t)DD * DD) return;
    const int n = (int)(i / DD), k = (int)(i % DD);
    Bt[(size_t)n * DD + k] = (_Float16)W[(size_t)k * DD + n];
}

// =====================================================================
// WMMA GEMM with LDS-staged weights:
//   C(M,N) = A(M,384) @ B(384,N) (+ bias[n])
// Each 256-thread block owns a 128-row x 64-col output tile:
//   - cooperatively copies the contiguous 64x384 f16 B strip into LDS
//     (48KB, padded stride -> bank-conflict-free ds_load_b128)
//   - each of the 8 waves computes one 16x64 strip: A fragment loaded
//     once per k-step (4x global_load_b128 + packed cvt) and reused by
//     4 WMMAs whose B fragments come from LDS.
// =====================================================================
__global__ __launch_bounds__(256) void k_wmma_gemm_lds(
    const float* __restrict__ A, const _Float16* __restrict__ Bt,
    const float* __restrict__ bias, float* __restrict__ C,
    int M, int N)
{
    __shared__ alignas(16) _Float16 sB[64 * SBS];   // ~50 KB

    const int Ng = N >> 6;                   // 64-col groups (6 for N=384)
    const int Mt = M >> 4;
    const int bm = blockIdx.x / Ng;          // block of 8 m-tiles
    const int tg = blockIdx.x % Ng;          // 64-col group

    // ---- cooperative copy: contiguous global strip -> padded LDS rows ----
    {
        const v8h* __restrict__ gsrc = (const v8h*)(Bt + (size_t)(tg << 6) * DD);
        const int nchunks = 64 * (DD / 8);   // 3072 x 16B
        for (int i = threadIdx.x; i < nchunks; i += 256) {
            const int r = i / (DD / 8), c = i % (DD / 8);
            *(v8h*)(sB + r * SBS + c * 8) = gsrc[(size_t)r * (DD / 8) + c];
        }
    }
    __syncthreads();

    const int wave = threadIdx.x >> 5;
    const int lane = threadIdx.x & 31;
    const int tm = bm * 8 + wave;
    if (tm < Mt) {                           // wave-uniform: EXEC all-ones inside
        const int half = (lane >> 4) << 3;   // 0 for lanes 0-15, 8 for 16-31
        const int mrow = (tm << 4) + (lane & 15);
        const int nl   = lane & 15;          // local column within the strip
        const float* __restrict__ Arow = A + (size_t)mrow * DD;

        v8f acc[4] = {{}, {}, {}, {}};
#pragma unroll 2
        for (int k0 = 0; k0 < DD; k0 += 32) {
            const int kb = k0 + half;
            // ---- A fragment: 4 x global_load_b128 + packed f32->f16 ----
            const float4 a0 = *(const float4*)(Arow + kb);
            const float4 a1 = *(const float4*)(Arow + kb + 4);
            const float4 a2 = *(const float4*)(Arow + kb + 16);
            const float4 a3 = *(const float4*)(Arow + kb + 20);
            v16h a;
            a[0]  = (_Float16)a0.x; a[1]  = (_Float16)a0.y;
            a[2]  = (_Float16)a0.z; a[3]  = (_Float16)a0.w;
            a[4]  = (_Float16)a1.x; a[5]  = (_Float16)a1.y;
            a[6]  = (_Float16)a1.z; a[7]  = (_Float16)a1.w;
            a[8]  = (_Float16)a2.x; a[9]  = (_Float16)a2.y;
            a[10] = (_Float16)a2.z; a[11] = (_Float16)a2.w;
            a[12] = (_Float16)a3.x; a[13] = (_Float16)a3.y;
            a[14] = (_Float16)a3.z; a[15] = (_Float16)a3.w;
            // ---- 4 B fragments from LDS (2 x ds_load_b128 each) ----
#pragma unroll
            for (int t = 0; t < 4; ++t) {
                const _Float16* Brow = sB + (nl + 16 * t) * SBS + kb;
                const v8h blo = *(const v8h*)(Brow);
                const v8h bhi = *(const v8h*)(Brow + 16);
                const v16h b = __builtin_shufflevector(
                    blo, bhi, 0, 1, 2, 3, 4, 5, 6, 7, 8, 9, 10, 11, 12, 13, 14, 15);
                acc[t] = __builtin_amdgcn_wmma_f32_16x16x32_f16(
                    false, a, false, b, (short)0, acc[t], false, false);
            }
        }

        const int mbase = (tm << 4) + half;  // C rows: v + (lane<16 ? 0 : 8)
#pragma unroll
        for (int t = 0; t < 4; ++t) {
            const int nc = (tg << 6) + nl + 16 * t;
            const float bv = bias ? bias[nc] : 0.0f;
#pragma unroll
            for (int v = 0; v < 8; ++v)
                C[(size_t)(mbase + v) * N + nc] = acc[t][v] + bv;
        }
    }
}

// ---------------- node embedding: h = x @ emb_W + emb_b ----------------
__global__ __launch_bounds__(DD) void k_node_linear(
    const float* __restrict__ x, const float* __restrict__ W,
    const float* __restrict__ b, float* __restrict__ h)
{
    __shared__ float xs[20];
    const int n = blockIdx.x, j = threadIdx.x;
    if (j < 20) xs[j] = x[n * 20 + j];
    __syncthreads();
    float acc = b[j];
#pragma unroll
    for (int k = 0; k < 20; ++k) acc += xs[k] * W[k * DD + j];
    h[(size_t)n * DD + j] = acc;
}

// ---------------- in-place LayerNorm (+ optional SiLU), row = 384 ----------------
__global__ __launch_bounds__(DD) void k_layernorm(
    float* __restrict__ X, const float* __restrict__ g,
    const float* __restrict__ b, int do_silu)
{
    const int j = threadIdx.x;
    const size_t base = (size_t)blockIdx.x * DD;
    float v = X[base + j];
    float s = v, q = v * v;
#pragma unroll
    for (int o = 16; o > 0; o >>= 1) {
        s += __shfl_xor(s, o, 32);
        q += __shfl_xor(q, o, 32);
    }
    __shared__ float ws[DD / 32], wq[DD / 32];
    const int wid = j >> 5;
    if ((j & 31) == 0) { ws[wid] = s; wq[wid] = q; }
    __syncthreads();
    float S = 0.f, Q = 0.f;
#pragma unroll
    for (int w = 0; w < DD / 32; ++w) { S += ws[w]; Q += wq[w]; }
    const float mu  = S * (1.0f / DD);
    const float var = Q * (1.0f / DD) - mu * mu;
    float y = (v - mu) * rsqrtf(var + 1e-5f) * g[j] + b[j];
    if (do_silu) y = silu_f(y);
    X[base + j] = y;
}

// ---------------- RBF + first edge linear + SiLU ----------------
__global__ __launch_bounds__(DD) void k_rbf_linear(
    const float* __restrict__ pos, const int* __restrict__ src,
    const int* __restrict__ dst, const float* __restrict__ W1,
    const float* __restrict__ b1, float* __restrict__ out)
{
    __shared__ float rb[NRBF];
    __shared__ float dsh;
    const int e = blockIdx.x, j = threadIdx.x;
    if (j == 0) {
        const int s = src[e], d = dst[e];
        const float dx = pos[s * 3 + 0] - pos[d * 3 + 0];
        const float dy = pos[s * 3 + 1] - pos[d * 3 + 1];
        const float dz = pos[s * 3 + 2] - pos[d * 3 + 2];
        dsh = sqrtf(dx * dx + dy * dy + dz * dz);
    }
    __syncthreads();
    if (j < NRBF) {
        const float spacing = 30.0f / (NRBF - 1);
        const float gamma = 1.0f / (spacing * spacing + 1e-8f);
        const float t = dsh - spacing * (float)j;
        rb[j] = __expf(-gamma * t * t);
    }
    __syncthreads();
    float acc = b1[j];
    for (int k = 0; k < NRBF; ++k) acc += rb[k] * W1[k * DD + j];
    out[(size_t)e * DD + j] = silu_f(acc);
}

// ---------------- GATv2 edge logits + segment max ----------------
__global__ __launch_bounds__(DD) void k_gat_logits(
    const float* __restrict__ xl, const float* __restrict__ xr,
    const float* __restrict__ ee, const float* __restrict__ att,
    const int* __restrict__ src, const int* __restrict__ dst,
    float* __restrict__ logits, unsigned* __restrict__ smax)
{
    const int e = blockIdx.x, j = threadIdx.x;
    const int s = src[e], d = dst[e];
    float v = xl[(size_t)s * DD + j] + xr[(size_t)d * DD + j] + ee[(size_t)e * DD + j];
    v = (v > 0.f) ? v : 0.2f * v;                 // leaky_relu(0.2)
    float t = v * att[j];
#pragma unroll
    for (int o = 16; o > 0; o >>= 1) t += __shfl_xor(t, o, 32);
    const int hd = j >> 5;                        // one wave per head
    if ((j & 31) == 0) {
        logits[(size_t)e * HH + hd] = t;
        atomicMax(&smax[(size_t)d * HH + hd], f2o(t));
    }
}

// ---------------- exp(logit - max) + segment sum; logits buffer becomes a ----------------
__global__ __launch_bounds__(256) void k_softmax_norm(
    float* __restrict__ lg, const unsigned* __restrict__ smax,
    const int* __restrict__ dst, float* __restrict__ denom)
{
    const size_t t = (size_t)blockIdx.x * blockDim.x + threadIdx.x;
    if (t >= (size_t)EE * HH) return;
    const int e = (int)(t / HH), hd = (int)(t % HH);
    const int d = dst[e];
    const float a = __expf(lg[t] - o2f(smax[(size_t)d * HH + hd]));
    lg[t] = a;
    atomicAdd(&denom[(size_t)d * HH + hd], a);
}

// ---------------- weighted message scatter (segment sum) ----------------
__global__ __launch_bounds__(DD) void k_scatter(
    const float* __restrict__ xl, const float* __restrict__ aw,
    const float* __restrict__ denom, const int* __restrict__ src,
    const int* __restrict__ dst, float* __restrict__ accum)
{
    const int e = blockIdx.x, j = threadIdx.x;
    const int s = src[e], d = dst[e];
    const int hd = j >> 5;
    const float w = aw[(size_t)e * HH + hd] / (denom[(size_t)d * HH + hd] + 1e-16f);
    atomicAdd(&accum[(size_t)d * DD + j], xl[(size_t)s * DD + j] * w);
}

// ---------------- bias + BatchNorm(eval) + SiLU + residual ----------------
__global__ __launch_bounds__(256) void k_node_update(
    const float* __restrict__ accum, const float* __restrict__ cb,
    const float* __restrict__ bg, const float* __restrict__ bb,
    const float* __restrict__ bm, const float* __restrict__ bv,
    float* __restrict__ h)
{
    const size_t i = (size_t)blockIdx.x * blockDim.x + threadIdx.x;
    if (i >= (size_t)NN * DD) return;
    const int j = (int)(i % DD);
    float o = accum[i] + cb[j];
    o = (o - bm[j]) * rsqrtf(bv[j] + 1e-5f) * bg[j] + bb[j];
    h[i] = silu_f(o) + h[i];
}

// ---------------- graph pooling accumulation ----------------
__global__ __launch_bounds__(DD) void k_pool_accum(
    const float* __restrict__ h, const int* __restrict__ batch,
    float* __restrict__ gmean, unsigned* __restrict__ gmax,
    float* __restrict__ cnt)
{
    const int n = blockIdx.x, j = threadIdx.x;
    const int g = batch[n];
    const float v = h[(size_t)n * DD + j];
    atomicAdd(&gmean[(size_t)g * DD + j], v);
    atomicMax(&gmax[(size_t)g * DD + j], f2o(v));
    if (j == 0) atomicAdd(&cnt[g], 1.0f);
}

// ---------------- per-graph MLP head ----------------
__global__ __launch_bounds__(DD) void k_head(
    const float* __restrict__ gmean, const unsigned* __restrict__ gmax,
    const float* __restrict__ cnt,
    const float* __restrict__ pW,  const float* __restrict__ pb,
    const float* __restrict__ hW1, const float* __restrict__ hb1,
    const float* __restrict__ hW2, const float* __restrict__ hb2,
    const float* __restrict__ hW3, const float* __restrict__ hb3,
    float* __restrict__ out)
{
    __shared__ float hg[2 * DD];
    __shared__ float buf[DD];
    const int g = blockIdx.x, j = threadIdx.x;
    const float c = fmaxf(cnt[g], 1.0f);
    hg[j]      = gmean[(size_t)g * DD + j] / c;
    hg[DD + j] = o2f(gmax[(size_t)g * DD + j]);
    __syncthreads();

    float a = pb[j];
    for (int k = 0; k < 2 * DD; ++k) a += hg[k] * pW[k * DD + j];
    buf[j] = silu_f(a);
    __syncthreads();

    float a2 = hb1[j];
    for (int k = 0; k < DD; ++k) a2 += buf[k] * hW1[k * DD + j];
    a2 = silu_f(a2);
    __syncthreads();
    buf[j] = a2;
    __syncthreads();

    float a3 = 0.0f;
    if (j < DD / 2) {
        a3 = hb2[j];
        for (int k = 0; k < DD; ++k) a3 += buf[k] * hW2[k * (DD / 2) + j];
        a3 = silu_f(a3);
    }
    __syncthreads();
    if (j < DD / 2) buf[j] = a3;
    __syncthreads();

    if (j == 0) {
        float o = hb3[0];
        for (int k = 0; k < DD / 2; ++k) o += buf[k] * hW3[k];
        out[g] = o;
    }
}

// =====================================================================
static inline void launch_gemm(const float* A, const _Float16* Bt, const float* bias,
                               float* C, int M, int N, hipStream_t s)
{
    const int Mt = M >> 4, Ng = N >> 6;
    const int blocks = ((Mt + 7) / 8) * Ng;    // 8 m-tiles per block
    k_wmma_gemm_lds<<<blocks, 256, 0, s>>>(A, Bt, bias, C, M, N);
}

extern "C" void kernel_launch(void* const* d_in, const int* in_sizes, int n_in,
                              void* d_out, int out_size, void* d_ws, size_t ws_size,
                              hipStream_t stream)
{
    (void)in_sizes; (void)n_in; (void)out_size; (void)ws_size;

    const float* x        = (const float*)d_in[0];
    const float* pos      = (const float*)d_in[1];
    const int*   ei       = (const int*)  d_in[2];
    const int*   batch    = (const int*)  d_in[3];
    const float* emb_W    = (const float*)d_in[4];
    const float* emb_b    = (const float*)d_in[5];
    const float* emb_g    = (const float*)d_in[6];
    const float* emb_beta = (const float*)d_in[7];
    const float* eW1      = (const float*)d_in[8];
    const float* eb1      = (const float*)d_in[9];
    const float* eW2      = (const float*)d_in[10];
    const float* eb2      = (const float*)d_in[11];
    const float* e_g      = (const float*)d_in[12];
    const float* e_beta   = (const float*)d_in[13];
    const float* Wl       = (const float*)d_in[14];
    const float* bl       = (const float*)d_in[15];
    const float* Wr       = (const float*)d_in[16];
    const float* br       = (const float*)d_in[17];
    const float* We       = (const float*)d_in[18];
    const float* att      = (const float*)d_in[19];
    const float* cb       = (const float*)d_in[20];
    const float* bn_g     = (const float*)d_in[21];
    const float* bn_b     = (const float*)d_in[22];
    const float* bn_m     = (const float*)d_in[23];
    const float* bn_v     = (const float*)d_in[24];
    const float* pW       = (const float*)d_in[25];
    const float* pb       = (const float*)d_in[26];
    const float* hW1      = (const float*)d_in[27];
    const float* hb1      = (const float*)d_in[28];
    const float* hW2      = (const float*)d_in[29];
    const float* hb2      = (const float*)d_in[30];
    const float* hW3      = (const float*)d_in[31];
    const float* hb3      = (const float*)d_in[32];

    const int* src = ei;
    const int* dst = ei + EE;

    // ---------------- workspace layout (floats) ----------------
    const size_t ND = (size_t)NN * DD;     // 7.68M
    const size_t ED = (size_t)EE * DD;     // 76.8M
    float* ws     = (float*)d_ws;
    float* h      = ws;                    // N x D
    float* xl     = h     + ND;            // N x D
    float* xr     = xl    + ND;            // N x D
    float* accum  = xr    + ND;            // N x D
    float* e      = accum + ND;            // E x D
    float* tmp1   = e     + ED;            // E x D (silu(rbf@eW1), reused as ee)
    float* logits = tmp1  + ED;            // E x H (reused as attention weights)
    float* smaxF  = logits + (size_t)EE * HH;   // N x H (used as unsigned)
    float* denom  = smaxF  + (size_t)NN * HH;   // N x H
    float* cnt    = denom  + (size_t)NN * HH;   // G
    float* gmean  = cnt + GG;              // G x D
    float* gmaxF  = gmean + (size_t)GG * DD;    // G x D (used as unsigned)
    _Float16* BtBase = (_Float16*)(gmaxF + (size_t)GG * DD); // 19 x D x D f16
    unsigned* smaxU = (unsigned*)smaxF;
    unsigned* gmaxU = (unsigned*)gmaxF;

    const size_t WSLOT = (size_t)DD * DD;  // elements per weight slot
    _Float16* eW2t = BtBase;               // slot 0
    _Float16* Wlt  = BtBase + WSLOT;       // slots 1..6
    _Float16* Wrt  = BtBase + 7 * WSLOT;   // slots 7..12
    _Float16* Wet  = BtBase + 13 * WSLOT;  // slots 13..18

    // ---------------- weight prep (f32 [K][N] -> f16 [N][K]) ----------------
    {
        const int blocks = (int)((WSLOT + 255) / 256);
        k_prep_bt<<<blocks, 256, 0, stream>>>(eW2, eW2t);
        for (int l = 0; l < LL; ++l) {
            k_prep_bt<<<blocks, 256, 0, stream>>>(Wl + l * WSLOT, Wlt + l * WSLOT);
            k_prep_bt<<<blocks, 256, 0, stream>>>(Wr + l * WSLOT, Wrt + l * WSLOT);
            k_prep_bt<<<blocks, 256, 0, stream>>>(We + l * WSLOT, Wet + l * WSLOT);
        }
    }

    // ---------------- node embedding ----------------
    k_node_linear<<<NN, DD, 0, stream>>>(x, emb_W, emb_b, h);
    k_layernorm<<<NN, DD, 0, stream>>>(h, emb_g, emb_beta, 1);

    // ---------------- edge encoder ----------------
    k_rbf_linear<<<EE, DD, 0, stream>>>(pos, src, dst, eW1, eb1, tmp1);
    launch_gemm(tmp1, eW2t, eb2, e, EE, DD, stream);
    k_layernorm<<<EE, DD, 0, stream>>>(e, e_g, e_beta, 0);

    // ---------------- GATv2 layers ----------------
    for (int l = 0; l < LL; ++l) {
        hipMemsetAsync(accum, 0, ND * sizeof(float), stream);
        hipMemsetAsync(smaxF, 0, (size_t)2 * NN * HH * sizeof(float), stream); // smax+denom

        launch_gemm(h, Wlt + l * WSLOT, bl + (size_t)l * DD, xl, NN, DD, stream);
        launch_gemm(h, Wrt + l * WSLOT, br + (size_t)l * DD, xr, NN, DD, stream);
        launch_gemm(e, Wet + l * WSLOT, nullptr, tmp1, EE, DD, stream);   // ee

        k_gat_logits<<<EE, DD, 0, stream>>>(xl, xr, tmp1, att + (size_t)l * HH * CC,
                                            src, dst, logits, smaxU);
        {
            const size_t tot = (size_t)EE * HH;
            k_softmax_norm<<<(unsigned)((tot + 255) / 256), 256, 0, stream>>>(
                logits, smaxU, dst, denom);
        }
        k_scatter<<<EE, DD, 0, stream>>>(xl, logits, denom, src, dst, accum);
        {
            const size_t tot = ND;
            k_node_update<<<(unsigned)((tot + 255) / 256), 256, 0, stream>>>(
                accum, cb + (size_t)l * DD,
                bn_g + (size_t)l * DD, bn_b + (size_t)l * DD,
                bn_m + (size_t)l * DD, bn_v + (size_t)l * DD, h);
        }
    }

    // ---------------- pooling + head ----------------
    hipMemsetAsync(cnt, 0, (GG + (size_t)2 * GG * DD) * sizeof(float), stream);
    k_pool_accum<<<NN, DD, 0, stream>>>(h, batch, gmean, gmaxU, cnt);
    k_head<<<GG, DD, 0, stream>>>(gmean, gmaxU, cnt, pW, pb,
                                  hW1, hb1, hW2, hb2, hW3, hb3, (float*)d_out);
}
